// KANNetwork_88098369175716
// MI455X (gfx1250) — compile-verified, matrix operand order checked by reading, not run on hardware
//
#include <hip/hip_runtime.h>
#include <hip/hip_bf16.h>

typedef __attribute__((ext_vector_type(16))) _Float16 v16h;
typedef __attribute__((ext_vector_type(8)))  float    v8f;

// ---------------------------------------------------------------------------
// Fast tanh via native exp (v_exp_f32). Clamp keeps exp finite in f32.
// ---------------------------------------------------------------------------
__device__ __forceinline__ float fast_tanh(float x) {
    float xc = fminf(fmaxf(x, -15.f), 15.f);
    float e  = __expf(2.f * xc);
    return (e - 1.f) / (e + 1.f);
}

// ---------------------------------------------------------------------------
// Column stats: mu and 1/(std_unbiased + 1e-6) per column.
// Block = 256 threads = 32 columns x 8 row-groups (coalesced column reads).
// ---------------------------------------------------------------------------
__global__ __launch_bounds__(256)
void colstats_kernel(const float* __restrict__ a, int rows, int cols,
                     float* __restrict__ mu, float* __restrict__ rstd) {
    __shared__ float ss[8][32];
    __shared__ float sq[8][32];
    int cl = threadIdx.x & 31;
    int g  = threadIdx.x >> 5;
    int c  = blockIdx.x * 32 + cl;
    float s = 0.f, q = 0.f;
    for (int r = g; r < rows; r += 8) {
        float v = a[(size_t)r * cols + c];
        s += v; q += v * v;
    }
    ss[g][cl] = s; sq[g][cl] = q;
    __syncthreads();
    if (threadIdx.x < 32) {
        float S = 0.f, Q = 0.f;
        #pragma unroll
        for (int g2 = 0; g2 < 8; ++g2) { S += ss[g2][threadIdx.x]; Q += sq[g2][threadIdx.x]; }
        float m   = S / (float)rows;
        float var = fmaxf((Q - S * m) / (float)(rows - 1), 0.f);
        float sd  = sqrtf(var);
        mu[c]   = m;
        rstd[c] = 1.f / (sd + 1e-6f);
    }
}

// ---------------------------------------------------------------------------
// f32 -> f16 cast. coeffs[o][i][n] flattened is already the [o][K] layout the
// WMMA B-fragment wants (K = i*16+n contiguous per output row) -> no transpose.
// ---------------------------------------------------------------------------
__global__ void cast_f16_kernel(const float* __restrict__ s,
                                _Float16* __restrict__ d, int n) {
    int i      = blockIdx.x * blockDim.x + threadIdx.x;
    int stride = gridDim.x * blockDim.x;
    for (; i < n; i += stride) d[i] = (_Float16)s[i];
}

// ---------------------------------------------------------------------------
// Fragment load from an LDS tile row (row-major, K contiguous per row).
// 16-bit A/B fragment layout (ISA 7.12.2): lanes 0-15 hold K {0..7,16..23},
// lanes 16-31 hold K {8..15,24..31} -> two ds_load_b128 per lane.
// ---------------------------------------------------------------------------
__device__ __forceinline__ v16h load_frag(const _Float16* row_ks, int hi) {
    union { v16h v; uint4 q[2]; } u;
    u.q[0] = *(const uint4*)(row_ks + hi * 8);
    u.q[1] = *(const uint4*)(row_ks + 16 + hi * 8);
    return u.v;
}

// ---------------------------------------------------------------------------
// KAN layer as WMMA GEMM with on-the-fly Gaussian RBF basis generation.
//   Out[b,o] = act( sum_k basis(Xin)[b,k] * Bg[o,k] ),  K = IN*16, N = 512.
// Block: 512 threads (16 waves), tile 256(M) x 128(N), K-step 64 (4 features),
// ping-pong double-buffered LDS (108 KB of 320 KB).
// RBF recurrence: e^{-2(x-c_n)^2} = e^{-2(x-c_0)^2} * t^n * w_n,
//   t = e^{(16/15)x}, w_n = e^{8-2c_n^2}  -> 2 exps + 16 muls per feature.
// ---------------------------------------------------------------------------
template<int IN, bool TANH>
__global__ __launch_bounds__(512)
void kan_gemm_kernel(const float* __restrict__ Xin,
                     const _Float16* __restrict__ Bg,
                     const float* __restrict__ mu,
                     const float* __restrict__ rstd,
                     float* __restrict__ Out) {
    constexpr int K   = IN * 16;
    constexpr int N   = 512;
    constexpr int LDT = 72;          // 64 + 8 f16 pad (row stride 144B, 16B-aligned)

    __shared__ _Float16 As[2][256][LDT];   // 73.7 KB
    __shared__ _Float16 Bs[2][128][LDT];   // 36.9 KB

    const int blockM = blockIdx.x * 256;
    const int blockN = blockIdx.y * 128;

    const int lane = threadIdx.x & 31;
    const int wid  = threadIdx.x >> 5;
    const int wm   = wid & 7;        // 8 M-subtiles of 32
    const int wn   = wid >> 3;       // 2 N-subtiles of 64
    const int tl   = lane & 15;
    const int hi   = lane >> 4;

    // per-n RBF constants, computed once and amortized over all K-steps
    float w[16];
    #pragma unroll
    for (int n = 0; n < 16; ++n) {
        float c = -2.f + (4.f / 15.f) * (float)n;
        w[n] = __expf(8.f - 2.f * c * c);
    }

    v8f acc[2][4];
    #pragma unroll
    for (int mf = 0; mf < 2; ++mf)
        #pragma unroll
        for (int nf = 0; nf < 4; ++nf) acc[mf][nf] = {};

    // ---- tile staging: B copy + A basis generation -------------------------
    auto stage = [&](int kk, int buf) {
        if (threadIdx.x < 256) {  // B: 128 rows x 64 k of f16, 64B per thread
            int n = threadIdx.x >> 1, part = threadIdx.x & 1;
            const uint4* src = (const uint4*)(Bg + (size_t)(blockN + n) * K + kk + 32 * part);
            uint4* dst = (uint4*)&Bs[buf][n][32 * part];
            dst[0] = src[0]; dst[1] = src[1]; dst[2] = src[2]; dst[3] = src[3];
        }
        {   // A: basis for 256 rows x 4 features; 2 features / thread
            int m = threadIdx.x >> 1, part = threadIdx.x & 1;
            int f0 = (kk >> 4) + 2 * part;
            const float2 v = *(const float2*)(Xin + (size_t)(blockM + m) * IN + f0);
            float vs[2] = { v.x, v.y };
            #pragma unroll
            for (int j = 0; j < 2; ++j) {
                int f = f0 + j;
                float xn = fminf(3.f, fmaxf(-3.f, (vs[j] - mu[f]) * rstd[f]));
                float d0 = xn + 2.f;                       // xn - c_0
                float p  = __expf(-2.f * d0 * d0);         // basis_0
                float t  = __expf((16.f / 15.f) * xn);     // ratio e^{4*delta*xn}
                _Float16* dst = &As[buf][m][(2 * part + j) * 16];
                #pragma unroll
                for (int n = 0; n < 16; ++n) {
                    dst[n] = (_Float16)(p * w[n]);
                    p *= t;
                }
            }
        }
    };

    stage(0, 0);
    __syncthreads();

    constexpr int NSTEPS = K / 64;
    for (int s = 0; s < NSTEPS; ++s) {
        const int buf = s & 1;
        if (s + 1 < NSTEPS) stage((s + 1) * 64, (s + 1) & 1);

        #pragma unroll
        for (int ks = 0; ks < 64; ks += 32) {
            v16h af[2], bf[4];
            #pragma unroll
            for (int mf = 0; mf < 2; ++mf)
                af[mf] = load_frag(&As[buf][wm * 32 + mf * 16 + tl][ks], hi);
            #pragma unroll
            for (int nf = 0; nf < 4; ++nf)
                bf[nf] = load_frag(&Bs[buf][wn * 64 + nf * 16 + tl][ks], hi);
            #pragma unroll
            for (int mf = 0; mf < 2; ++mf)
                #pragma unroll
                for (int nf = 0; nf < 4; ++nf)
                    acc[mf][nf] = __builtin_amdgcn_wmma_f32_16x16x32_f16(
                        false, af[mf], false, bf[nf],
                        (short)0, acc[mf][nf], false, false);
        }
        __syncthreads();
    }

    // ---- store (C/D layout: vgpr r -> M=r (lanes 0-15) / M=8+r (16-31)) ----
    #pragma unroll
    for (int mf = 0; mf < 2; ++mf) {
        #pragma unroll
        for (int nf = 0; nf < 4; ++nf) {
            int rbase = blockM + wm * 32 + mf * 16 + hi * 8;
            int col   = blockN + wn * 64 + nf * 16 + tl;
            #pragma unroll
            for (int r = 0; r < 8; ++r) {
                float v = acc[mf][nf][r];
                if (TANH) v = fast_tanh(v);
                Out[(size_t)(rbase + r) * N + col] = v;
            }
        }
    }
}

// ---------------------------------------------------------------------------
// Layer 3 (GEMV over basis of h2) + skip connection. One wave per batch row.
// ---------------------------------------------------------------------------
__global__ __launch_bounds__(256)
void kan_final_kernel(const float* __restrict__ h2,
                      const float* __restrict__ mu,
                      const float* __restrict__ rstd,
                      const float* __restrict__ c3,
                      const float* __restrict__ x,
                      const float* __restrict__ skip_w,
                      const float* __restrict__ skip_b,
                      float* __restrict__ out) {
    const int row  = blockIdx.x * 8 + (threadIdx.x >> 5);
    const int lane = threadIdx.x & 31;
    float w[16];
    #pragma unroll
    for (int n = 0; n < 16; ++n) {
        float c = -2.f + (4.f / 15.f) * (float)n;
        w[n] = __expf(8.f - 2.f * c * c);
    }
    float acc = 0.f;
    for (int i = lane; i < 512; i += 32) {
        float v  = h2[(size_t)row * 512 + i];
        float xn = fminf(3.f, fmaxf(-3.f, (v - mu[i]) * rstd[i]));
        float d0 = xn + 2.f;
        float p  = __expf(-2.f * d0 * d0);
        float t  = __expf((16.f / 15.f) * xn);
        const float* cc = c3 + i * 16;
        #pragma unroll
        for (int n = 0; n < 16; ++n) {
            acc += (p * w[n]) * cc[n];
            p *= t;
        }
    }
    for (int j = lane; j < 256; j += 32)
        acc += x[(size_t)row * 256 + j] * skip_w[j];
    #pragma unroll
    for (int off = 16; off; off >>= 1) acc += __shfl_xor(acc, off, 32);
    if (lane == 0) out[row] = acc + skip_b[0];
}

// ---------------------------------------------------------------------------
extern "C" void kernel_launch(void* const* d_in, const int* in_sizes, int n_in,
                              void* d_out, int out_size, void* d_ws, size_t ws_size,
                              hipStream_t stream) {
    const float* x  = (const float*)d_in[0];  // [16384, 256]
    const float* c1 = (const float*)d_in[1];  // [512, 256, 16]
    const float* c2 = (const float*)d_in[2];  // [512, 512, 16]
    const float* c3 = (const float*)d_in[3];  // [1, 512, 16]
    const float* sw = (const float*)d_in[4];  // [1, 256]
    const float* sb = (const float*)d_in[5];  // [1]
    float* out = (float*)d_out;               // [16384]

    constexpr int B = 16384, IN = 256, H = 512;
    constexpr size_t K1 = (size_t)IN * 16;    // 4096
    constexpr size_t K2 = (size_t)H * 16;     // 8192

    char* ws = (char*)d_ws;
    size_t off = 0;
    auto alloc = [&](size_t bytes) { void* p = ws + off; off = (off + bytes + 255) & ~(size_t)255; return p; };
    _Float16* c1h = (_Float16*)alloc(K1 * H * sizeof(_Float16));  // 4 MiB
    _Float16* c2h = (_Float16*)alloc(K2 * H * sizeof(_Float16));  // 8 MiB
    float* h1    = (float*)alloc((size_t)B * H * sizeof(float));  // 32 MiB
    float* h2    = (float*)alloc((size_t)B * H * sizeof(float));  // 32 MiB
    float* mu0   = (float*)alloc(IN * sizeof(float));
    float* rs0   = (float*)alloc(IN * sizeof(float));
    float* mu1   = (float*)alloc(H * sizeof(float));
    float* rs1   = (float*)alloc(H * sizeof(float));
    float* mu2   = (float*)alloc(H * sizeof(float));
    float* rs2   = (float*)alloc(H * sizeof(float));
    (void)ws_size; (void)in_sizes; (void)n_in; (void)out_size;

    // coeff casts (B matrices are already [o][K] contiguous -> pure cast)
    cast_f16_kernel<<<1024, 256, 0, stream>>>(c1, c1h, (int)(K1 * H));
    cast_f16_kernel<<<1024, 256, 0, stream>>>(c2, c2h, (int)(K2 * H));

    // layer 1
    colstats_kernel<<<IN / 32, 256, 0, stream>>>(x, B, IN, mu0, rs0);
    kan_gemm_kernel<IN, true><<<dim3(B / 256, H / 128), 512, 0, stream>>>(x, c1h, mu0, rs0, h1);

    // layer 2
    colstats_kernel<<<H / 32, 256, 0, stream>>>(h1, B, H, mu1, rs1);
    kan_gemm_kernel<H, true><<<dim3(B / 256, H / 128), 512, 0, stream>>>(h1, c2h, mu1, rs1, h2);

    // layer 3 + skip
    colstats_kernel<<<H / 32, 256, 0, stream>>>(h2, B, H, mu2, rs2);
    kan_final_kernel<<<B / 8, 256, 0, stream>>>(h2, mu2, rs2, c3, x, sw, sb, out);
}
